// MambaStateSpaceModel_48378511622291
// MI455X (gfx1250) — matrix operand order repaired and checked by reading, not compiled, not run on hardware
//
#include <hip/hip_runtime.h>
#include <math.h>

#define SEQ   8192
#define INDIM 1024
#define HID   2048
#define RGRID (HID / 16)      // 128 persistent workgroups, 16 rows each
#define APAD  2056            // 2048 + 8 bf16 pad to spread LDS banks

typedef __attribute__((ext_vector_type(16))) __bf16 v16bf;
typedef __attribute__((ext_vector_type(8)))  __bf16 v8bf;
typedef __attribute__((ext_vector_type(8)))  float  v8f;
typedef __attribute__((ext_vector_type(4)))  int    v4i;

#if defined(__has_builtin)
#if __has_builtin(__builtin_amdgcn_global_load_async_to_lds_b128) && \
    __has_builtin(__builtin_amdgcn_s_wait_asynccnt)
#define HAVE_ASYNC_LDS 1
#endif
#endif

#if defined(HAVE_ASYNC_LDS)
typedef __attribute__((address_space(1))) v4i gv4i;   // global (device) AS
typedef __attribute__((address_space(3))) v4i lv4i;   // LDS AS
#endif

// ---------- helpers: pack WMMA fragments per CDNA5 ISA 7.12.2 layouts ----------
// A-operand (16x32 bf16): lanes 0-15 -> M=lane, elems 0..7 = K[base..base+7],
// elems 8..15 = K[base+16..base+23]; lanes 16-31 same with base+8 shift.
__device__ inline v16bf pack_a(const __bf16* rowp, int k0, bool hi) {
  const __bf16* p = rowp + k0 + (hi ? 8 : 0);
  v8bf lo = *(const v8bf*)(p);
  v8bf hh = *(const v8bf*)(p + 16);
  v16bf a;
#pragma unroll
  for (int i = 0; i < 8; ++i) { a[i] = lo[i]; a[8 + i] = hh[i]; }
  return a;
}
// B-operand (32x16 bf16): lanes 0-15 -> N=lane, elems 0..7 = K[k0..k0+7],
// elems 8..15 = K[k0+16..k0+23]; lanes 16-31 same with k0+8 shift.
__device__ inline v16bf pack_b(const __bf16* colp, int k0, bool hi) {
  const __bf16* p = colp + k0 + (hi ? 8 : 0);
  v8bf lo = *(const v8bf*)(p);
  v8bf hh = *(const v8bf*)(p + 16);
  v16bf b;
#pragma unroll
  for (int i = 0; i < 8; ++i) { b[i] = lo[i]; b[8 + i] = hh[i]; }
  return b;
}

// ---------- init: zero sync counter and h buffers (runs every launch) ----------
__global__ void init_state(unsigned* cnt, __bf16* hbf, float* hf32) {
  int i = blockIdx.x * blockDim.x + threadIdx.x;
  if (i == 0) *cnt = 0u;
  if (i < 2 * HID) hbf[i] = (__bf16)0.0f;   // both ping-pong buffers
  if (i < HID) hf32[i] = 0.0f;
}

// ---------- fp32 -> bf16 conversion ----------
__global__ void cvt_f32_bf16(const float* __restrict__ src, __bf16* __restrict__ dst, int n) {
  int i = blockIdx.x * blockDim.x + threadIdx.x;
  if (i < n) dst[i] = (__bf16)src[i];
}

// ---------- phase 1: Bx = x @ B^T + bias via v_wmma_f32_16x16x32_bf16 ----------
// Each wave computes a 16(M) x 64(N) strip: one A-fragment feeds 4 WMMAs.
__global__ void gemm_bx(const __bf16* __restrict__ xb, const __bf16* __restrict__ Bb,
                        const float* __restrict__ bias, float* __restrict__ Bx) {
  const int lane = threadIdx.x & 31;
  const int w    = blockIdx.x * 8 + (threadIdx.x >> 5);
  const int NT64 = HID / 64;            // 32 n-strips
  const int m0   = (w / NT64) * 16;
  const int n0   = (w % NT64) * 64;
  const bool hi  = lane >= 16;
  const int lr   = lane & 15;

  const __bf16* arow = xb + (size_t)(m0 + lr) * INDIM;   // x row (A operand, M=lr)
  const __bf16* brow0 = Bb + (size_t)(n0 + lr) * INDIM;  // B rows n -> columns of B^T
  const __bf16* brow1 = brow0 + (size_t)16 * INDIM;
  const __bf16* brow2 = brow0 + (size_t)32 * INDIM;
  const __bf16* brow3 = brow0 + (size_t)48 * INDIM;

  v8f acc0 = {}, acc1 = {}, acc2 = {}, acc3 = {};
#pragma unroll 2
  for (int c = 0; c < INDIM / 32; ++c) {
    const int k0 = c * 32;
    v16bf a  = pack_a(arow, k0, hi);
    v16bf b0 = pack_b(brow0, k0, hi);
    v16bf b1 = pack_b(brow1, k0, hi);
    v16bf b2 = pack_b(brow2, k0, hi);
    v16bf b3 = pack_b(brow3, k0, hi);
    acc0 = __builtin_amdgcn_wmma_f32_16x16x32_bf16(false, a, false, b0, (short)0, acc0, false, false);
    acc1 = __builtin_amdgcn_wmma_f32_16x16x32_bf16(false, a, false, b1, (short)0, acc1, false, false);
    acc2 = __builtin_amdgcn_wmma_f32_16x16x32_bf16(false, a, false, b2, (short)0, acc2, false, false);
    acc3 = __builtin_amdgcn_wmma_f32_16x16x32_bf16(false, a, false, b3, (short)0, acc3, false, false);
  }
  const int rb = m0 + (hi ? 8 : 0);     // D layout: vgpr r -> M = rb + r, N = lr
#pragma unroll
  for (int s = 0; s < 4; ++s) {
    v8f acc = (s == 0) ? acc0 : (s == 1) ? acc1 : (s == 2) ? acc2 : acc3;
    const int   n  = n0 + s * 16 + lr;
    const float bs = bias[n];
#pragma unroll
    for (int r = 0; r < 8; ++r)
      Bx[(size_t)(rb + r) * HID + n] = acc[r] + bs;
  }
}

// ---------- phase 2: persistent recurrence, A pinned in LDS, grid barrier ----------
// 128 blocks x 256 threads. Block b owns rows [16b, 16b+16). 8 waves split K 8 ways.
__global__ void recurrence(const __bf16* __restrict__ Ab, const float* __restrict__ Bx,
                           __bf16* __restrict__ hbf, float* __restrict__ hf32,
                           unsigned* __restrict__ cnt) {
  extern __shared__ char smem[];
  __bf16* As      = (__bf16*)smem;                               // 16 x APAD bf16
  __bf16* hs      = (__bf16*)(smem + 16 * APAD * 2);             // 2048 bf16
  float*  partial = (float*) (smem + 16 * APAD * 2 + HID * 2);   // 8 x 16 f32

  const int tid  = threadIdx.x;
  const int lane = tid & 31;
  const int wv   = tid >> 5;          // 0..7 : K-split
  const bool hi  = lane >= 16;
  const int lr   = lane & 15;
  const int r0   = blockIdx.x * 16;
  const unsigned G = gridDim.x;

  // Preload this block's 16 rows of A into LDS (bf16, padded stride).
  for (int idx = tid; idx < 16 * (HID / 8); idx += blockDim.x) {
    int row = idx >> 8;               // HID/8 = 256 chunks per row
    int col = (idx & 255) << 3;
    *(v8bf*)(As + row * APAD + col) =
        *(const v8bf*)(Ab + (size_t)(r0 + row) * HID + col);
  }
  __syncthreads();

  const __bf16* arow = As + lr * APAD;
  const int kwv = wv * (HID / 8);     // 256-wide K band per wave

  for (int t = 0; t < SEQ; ++t) {
    // stage read-buffer h into LDS (double buffered by t parity)
    const __bf16* hrd = hbf + (t & 1) * HID;
#if defined(HAVE_ASYNC_LDS)
    {
      // global_load_async_to_lds_b128: (AS1 int4*, AS3 int4*, imm offset, cpol)
      __builtin_amdgcn_global_load_async_to_lds_b128(
          (gv4i*)(hrd + tid * 8),
          (lv4i*)(hs + tid * 8), 0, 0);
      __builtin_amdgcn_s_wait_asynccnt(0);
    }
#else
    *(v8bf*)(hs + tid * 8) = *(const v8bf*)(hrd + tid * 8);
#endif
    __syncthreads();

    v8f acc = {};
#pragma unroll
    for (int c = 0; c < 8; ++c) {
      int k0 = kwv + c * 32;
      v16bf a = pack_a(arow, k0, hi);
      v16bf b = pack_b(hs, k0, hi);   // h replicated across all 16 N columns
      acc = __builtin_amdgcn_wmma_f32_16x16x32_bf16(false, a, false, b, (short)0, acc, false, false);
    }
    // column N=0 lives in lane 0 (M=0..7) and lane 16 (M=8..15)
    if (lr == 0) {
      int ro = hi ? 8 : 0;
#pragma unroll
      for (int r = 0; r < 8; ++r) partial[wv * 16 + ro + r] = acc[r];
    }
    __syncthreads();

    if (tid < 16) {
      float s = 0.0f;
#pragma unroll
      for (int w = 0; w < 8; ++w) s += partial[w * 16 + tid];
      s += Bx[(size_t)t * HID + r0 + tid];
      float hn = tanhf(s);
      hbf[((t + 1) & 1) * HID + r0 + tid] = (__bf16)hn;
      if (t == SEQ - 1) hf32[r0 + tid] = hn;
      __threadfence();                 // make h slice visible before arrive
    }
    __syncthreads();

    if (tid == 0) {
      __hip_atomic_fetch_add(cnt, 1u, __ATOMIC_RELEASE, __HIP_MEMORY_SCOPE_AGENT);
      const unsigned target = G * (unsigned)(t + 1);   // monotonic: replay-safe
      while (__hip_atomic_load(cnt, __ATOMIC_ACQUIRE, __HIP_MEMORY_SCOPE_AGENT) < target)
        __builtin_amdgcn_s_sleep(4);
    }
    __syncthreads();
  }
}

// ---------- phase 3: out = W_out . h_final + b_out ----------
__global__ void final_out(const float* __restrict__ W, const float* __restrict__ b,
                          const float* __restrict__ hf32, float* __restrict__ out) {
  __shared__ float red[256];
  float s = 0.0f;
  for (int i = threadIdx.x; i < HID; i += 256) s += W[i] * hf32[i];
  red[threadIdx.x] = s;
  __syncthreads();
  for (int st = 128; st > 0; st >>= 1) {
    if (threadIdx.x < st) red[threadIdx.x] += red[threadIdx.x + st];
    __syncthreads();
  }
  if (threadIdx.x == 0) out[0] = red[0] + b[0];
}

extern "C" void kernel_launch(void* const* d_in, const int* in_sizes, int n_in,
                              void* d_out, int out_size, void* d_ws, size_t ws_size,
                              hipStream_t stream) {
  const float* x    = (const float*)d_in[0];   // (8192,1024)
  const float* A    = (const float*)d_in[1];   // (2048,2048)
  const float* B    = (const float*)d_in[2];   // (2048,1024)
  const float* bias = (const float*)d_in[3];   // (2048,)
  const float* Wout = (const float*)d_in[4];   // (1,2048)
  const float* bout = (const float*)d_in[5];   // (1,)
  float* out = (float*)d_out;

  char* ws = (char*)d_ws;
  unsigned* cnt  = (unsigned*)(ws);                       // sync counter
  __bf16*   hbf  = (__bf16*)(ws + 1024);                  // 2 x 2048 bf16 (ping-pong)
  float*    hf32 = (float*)(ws + 16 * 1024);              // 2048 f32
  __bf16*   xb   = (__bf16*)(ws + 64 * 1024);             // 16 MB
  __bf16*   Bb   = xb + (size_t)SEQ * INDIM;              // 4 MB
  __bf16*   Ab   = Bb + (size_t)HID * INDIM;              // 8 MB
  float*    Bx   = (float*)(Ab + (size_t)HID * HID);      // 64 MB

  init_state<<<(2 * HID + 255) / 256, 256, 0, stream>>>(cnt, hbf, hf32);
  cvt_f32_bf16<<<(SEQ * INDIM) / 256, 256, 0, stream>>>(x, xb, SEQ * INDIM);
  cvt_f32_bf16<<<(HID * INDIM) / 256, 256, 0, stream>>>(B, Bb, HID * INDIM);
  cvt_f32_bf16<<<(HID * HID)  / 256, 256, 0, stream>>>(A, Ab, HID * HID);

  gemm_bx<<<(SEQ / 16) * (HID / 64) / 8, 256, 0, stream>>>(xb, Bb, bias, Bx);

  size_t smem = (size_t)16 * APAD * 2 + (size_t)HID * 2 + 8 * 16 * 4;  // ~70.4 KB
  recurrence<<<RGRID, 256, smem, stream>>>(Ab, Bx, hbf, hf32, cnt);

  final_out<<<1, 256, 0, stream>>>(Wout, bout, hf32, out);
}
